// LSTMDecoder_37288906064634
// MI455X (gfx1250) — matrix operand order repaired and checked by reading, not compiled, not run on hardware
//
#include <hip/hip_runtime.h>
#include <hip/hip_bf16.h>

typedef __bf16 v16bf __attribute__((ext_vector_type(16)));
typedef float  v8f   __attribute__((ext_vector_type(8)));

union BF16x16 { uint4 q[2]; v16bf v; };

__device__ inline unsigned short f2bf(float f) {
    unsigned int u = __float_as_uint(f);
    unsigned int r = u + 0x7FFFu + ((u >> 16) & 1u);   // round-to-nearest-even
    return (unsigned short)(r >> 16);
}
__device__ inline v8f v8f_zero() { v8f z; for (int i = 0; i < 8; ++i) z[i] = 0.0f; return z; }
__device__ inline float sigmoidf(float x) { return 1.0f / (1.0f + __expf(-x)); }

__device__ inline v8f wmma_bf16(v16bf a, v16bf b, v8f c) {
    return __builtin_amdgcn_wmma_f32_16x16x32_bf16(false, a, false, b, (short)0, c, false, false);
}

// ---- constants ----
#define B_   64
#define S_   512
#define H_   512
#define P_   128
#define G_   2048   // 4*H
#define INX  14

// ---------------------------------------------------------------------------
// W_eff[g][k] = sum_{j<512} W_ih[g][j] * Wemb[j][k]   (Wemb = [cmd_W | coord_W])
__global__ __launch_bounds__(256) void prep_weff(const float* __restrict__ W_ih,
                                                 const float* __restrict__ cmdW,
                                                 const float* __restrict__ coordW,
                                                 float* __restrict__ weff) {
    int idx = blockIdx.x * 256 + threadIdx.x;         // 2048*16
    int g = idx >> 4, k = idx & 15;
    float s = 0.0f;
    const float* wr = W_ih + (size_t)g * 768;
    if (k < 6) {
        for (int j = 0; j < 512; ++j) s += wr[j] * cmdW[j * 6 + k];
    } else if (k < 14) {
        int kk = k - 6;
        for (int j = 0; j < 512; ++j) s += wr[j] * coordW[j * 8 + kk];
    }
    weff[(size_t)g * 16 + k] = s;
}

// base[b][g] = b_ih+b_hh + W_ih1 @ (cmd_b+coord_b) + W_ih2 @ ctx[b]
__global__ __launch_bounds__(256) void prep_base(const float* __restrict__ W_ih,
                                                 const float* __restrict__ ctx,
                                                 const float* __restrict__ cmdb,
                                                 const float* __restrict__ coordb,
                                                 const float* __restrict__ b_ih,
                                                 const float* __restrict__ b_hh,
                                                 float* __restrict__ base) {
    int idx = blockIdx.x * 256 + threadIdx.x;         // 64*2048
    int b = idx >> 11, g = idx & 2047;
    const float* wr = W_ih + (size_t)g * 768;
    float s = b_ih[g] + b_hh[g];
    for (int j = 0; j < 512; ++j) s += wr[j] * (cmdb[j] + coordb[j]);
    const float* cx = ctx + (size_t)b * 256;
    for (int j = 0; j < 256; ++j) s += wr[512 + j] * cx[j];
    base[(size_t)b * G_ + g] = s;
}

// convert W_hh (2048x128) and W_hr (128x512) to bf16
__global__ __launch_bounds__(256) void prep_cvt(const float* __restrict__ W_hh,
                                                const float* __restrict__ W_hr,
                                                unsigned short* __restrict__ whh_bf,
                                                unsigned short* __restrict__ whr_bf) {
    int idx = blockIdx.x * 256 + threadIdx.x;         // 262144 + 65536
    if (idx < G_ * P_) whh_bf[idx] = f2bf(W_hh[idx]);
    else               whr_bf[idx - G_ * P_] = f2bf(W_hr[idx - G_ * P_]);
}

// Precomputed input-gates, stored in f32 WMMA C-fragment order:
// xgf[(((t*4 + bt)*128 + ntile)*32 + lane)*8 + r]
//   = base + x@W_eff.T  at  batch row b0 + (lane>>4)*8 + r, gate col ntile*16 + (lane&15)
__global__ __launch_bounds__(256) void xgates_kernel(const float* __restrict__ x,
                                                     const float* __restrict__ weff,
                                                     const float* __restrict__ base,
                                                     float* __restrict__ xgf) {
    __shared__ float xs[16][16];                  // [mloc][k] for this batch tile
    int ntile = blockIdx.x, bt = blockIdx.y, s = blockIdx.z;
    int b0 = bt * 16;
    int tid = threadIdx.x;
    if (tid < 16 * INX) {
        int mm = tid / INX, kk = tid % INX;
        xs[mm][kk] = x[((size_t)(b0 + mm) * S_ + s) * INX + kk];
    }
    __syncthreads();
    int lane = tid >> 3, r = tid & 7;
    int mloc = (lane >> 4) * 8 + r;
    int g = ntile * 16 + (lane & 15);
    const float* w = weff + (size_t)g * 16;
    float v = base[(size_t)(b0 + mloc) * G_ + g];
    #pragma unroll
    for (int k = 0; k < INX; ++k) v += xs[mloc][k] * w[k];
    xgf[((((size_t)s * 4 + bt) * 128 + ntile) * 32 + lane) * 8 + r] = v;   // coalesced
}

// ---------------------------------------------------------------------------
// Persistent recurrence: 4 workgroups x 256 threads (8 waves), batch tile 16.
// Low occupancy (2 waves/SIMD) buys a ~256-VGPR/wave budget so a deep prefetch
// pipeline survives register allocation. Wave w owns h-slice [w*64, w*64+64):
// 4 n-chunks x 4 gate types. B tiles (W_hh/W_hr, L2-resident) are fetched in
// 2-group ring: each 8-load clause is issued one kc-group (4 WMMAs) ahead.
// Phase B uses all 8 waves (one 16-col p-tile each).
__global__ __launch_bounds__(256, 1) void scan_kernel(const float* __restrict__ xgf,
                                                      const unsigned short* __restrict__ whh,
                                                      const unsigned short* __restrict__ whr,
                                                      float* __restrict__ hs) {
    __shared__ unsigned short h_buf[16][128];   // h_{t-1}, bf16, [m][k]
    __shared__ unsigned short a_buf[16][512];   // o*tanh(c), bf16, [m][k]

    const int tid   = threadIdx.x;
    const int wave  = tid >> 5;                 // 0..7
    const int lane  = tid & 31;
    const int bt    = blockIdx.x;
    const int b0    = bt * 16;
    const int mhalf = lane >> 4;                // lane half (0/1)
    const int nloc  = lane & 15;
    const int hbase = wave * 64;                // this wave's h-slice
    const int arow  = lane & 15;                // A-operand: lane = M row
    const int koff  = mhalf * 16;               // A-operand K sub-offset

    for (int i = tid; i < 16 * 128; i += 256) ((unsigned short*)h_buf)[i] = 0;
    __syncthreads();

    v8f cst[4];                                 // cell state, C/D tile layout
    #pragma unroll
    for (int i = 0; i < 4; ++i) cst[i] = v8f_zero();

    #pragma unroll 1
    for (int t = 0; t < S_; ++t) {
        // ---- phase A: gates = xg_t + h @ W_hh.T ; cell update ----
        #pragma unroll 1
        for (int n2 = 0; n2 < 4; ++n2) {
            const int ncol = hbase + n2 * 16 + nloc;      // per-lane N folded in
            // C init: contiguous 32B v8f fragment loads (i,f,g,o of this chunk)
            const float* cf =
                xgf + ((((size_t)t * 4 + bt) * 128 + wave * 4 + n2) * 32 + lane) * 8;
            v8f acc[4];
            #pragma unroll
            for (int q = 0; q < 4; ++q)
                acc[q] = *reinterpret_cast<const v8f*>(cf + (size_t)q * 8192);

            // kc loop with a 2-group B ring: group kc+1 loads while group kc
            // feeds 4 WMMAs. 8 tiles (64 VGPRs) in flight.
            BF16x16 Bg[2][4];
            BF16x16 A[2];
            #pragma unroll
            for (int q = 0; q < 4; ++q) {                  // preload group kc=0
                const unsigned short* bp =
                    whh + (size_t)(q * 512 + ncol) * P_ + mhalf * 16;
                Bg[0][q].q[0] = *reinterpret_cast<const uint4*>(bp);
                Bg[0][q].q[1] = *reinterpret_cast<const uint4*>(bp + 8);
            }
            #pragma unroll
            for (int kc = 0; kc < 4; ++kc) {
                const int k0 = kc * 32;
                if (kc < 3) {                              // prefetch group kc+1
                    const int kn = (kc + 1) * 32;
                    #pragma unroll
                    for (int q = 0; q < 4; ++q) {
                        const unsigned short* bp =
                            whh + (size_t)(q * 512 + ncol) * P_ + kn + mhalf * 16;
                        Bg[(kc + 1) & 1][q].q[0] = *reinterpret_cast<const uint4*>(bp);
                        Bg[(kc + 1) & 1][q].q[1] = *reinterpret_cast<const uint4*>(bp + 8);
                    }
                }
                A[kc & 1].q[0] = *reinterpret_cast<const uint4*>(&h_buf[arow][k0 + koff]);
                A[kc & 1].q[1] = *reinterpret_cast<const uint4*>(&h_buf[arow][k0 + koff + 8]);
                #pragma unroll
                for (int q = 0; q < 4; ++q)
                    acc[q] = wmma_bf16(A[kc & 1].v, Bg[kc & 1][q].v, acc[q]);
            }
            // cell update for this chunk; consumes acc immediately
            #pragma unroll
            for (int r = 0; r < 8; ++r) {
                float ig = sigmoidf(acc[0][r]);
                float fg = sigmoidf(acc[1][r]);
                float gg = tanhf(acc[2][r]);
                float og = sigmoidf(acc[3][r]);
                float cn = fg * cst[n2][r] + ig * gg;
                cst[n2][r] = cn;
                float av = og * tanhf(cn);
                int M = mhalf * 8 + r;
                a_buf[M][hbase + n2 * 16 + nloc] = f2bf(av);
            }
        }
        __syncthreads();

        // ---- phase B: h = a @ W_hr.T  (all 8 waves, one 16-col p-tile each) ----
        {
            const int p0 = wave * 16;
            v8f h = v8f_zero();
            const unsigned short* bbase = whr + (size_t)(p0 + nloc) * H_ + mhalf * 16;
            BF16x16 Bg[2][4];
            BF16x16 Ab[2];
            #pragma unroll
            for (int i = 0; i < 4; ++i) {                  // preload group kg=0
                const unsigned short* bp = bbase + i * 32;
                Bg[0][i].q[0] = *reinterpret_cast<const uint4*>(bp);
                Bg[0][i].q[1] = *reinterpret_cast<const uint4*>(bp + 8);
            }
            #pragma unroll
            for (int kg = 0; kg < 4; ++kg) {
                if (kg < 3) {                              // prefetch group kg+1
                    #pragma unroll
                    for (int i = 0; i < 4; ++i) {
                        const unsigned short* bp = bbase + ((kg + 1) * 4 + i) * 32;
                        Bg[(kg + 1) & 1][i].q[0] = *reinterpret_cast<const uint4*>(bp);
                        Bg[(kg + 1) & 1][i].q[1] = *reinterpret_cast<const uint4*>(bp + 8);
                    }
                }
                #pragma unroll
                for (int i = 0; i < 4; ++i) {
                    const int k0 = (kg * 4 + i) * 32;
                    Ab[i & 1].q[0] = *reinterpret_cast<const uint4*>(&a_buf[arow][k0 + koff]);
                    Ab[i & 1].q[1] = *reinterpret_cast<const uint4*>(&a_buf[arow][k0 + koff + 8]);
                    h = wmma_bf16(Ab[i & 1].v, Bg[kg & 1][i].v, h);
                }
            }
            #pragma unroll
            for (int r = 0; r < 8; ++r) {
                int M = mhalf * 8 + r;
                h_buf[M][p0 + nloc] = f2bf(h[r]);
                hs[(((size_t)(b0 + M)) * S_ + t) * P_ + p0 + nloc] = h[r];
            }
        }
        __syncthreads();
    }
}

// ---------------------------------------------------------------------------
// LayerNorm + two small heads. One wave per (b,s) token.
__global__ __launch_bounds__(256) void out_kernel(const float* __restrict__ hs,
                                                  const float* __restrict__ ln_g,
                                                  const float* __restrict__ ln_b,
                                                  const float* __restrict__ outcW,
                                                  const float* __restrict__ outcb,
                                                  const float* __restrict__ outxW,
                                                  const float* __restrict__ outxb,
                                                  const float* __restrict__ scale_p,
                                                  float* __restrict__ out) {
    __shared__ float y_s[8][128];
    __shared__ float cmd_s[8][6];
    int wave = threadIdx.x >> 5, lane = threadIdx.x & 31;
    int idx = blockIdx.x * 8 + wave;               // b*512 + s, exact cover
    const float* hp = hs + (size_t)idx * P_;

    float v[4]; float sum = 0.0f;
    #pragma unroll
    for (int j = 0; j < 4; ++j) { v[j] = hp[lane * 4 + j]; sum += v[j]; }
    for (int m = 16; m >= 1; m >>= 1) sum += __shfl_xor(sum, m, 32);
    float mu = sum * (1.0f / 128.0f);
    float s2 = 0.0f;
    #pragma unroll
    for (int j = 0; j < 4; ++j) { float d = v[j] - mu; s2 += d * d; }
    for (int m = 16; m >= 1; m >>= 1) s2 += __shfl_xor(s2, m, 32);
    float rstd = rsqrtf(s2 * (1.0f / 128.0f) + 1e-5f);
    #pragma unroll
    for (int j = 0; j < 4; ++j) {
        int p = lane * 4 + j;
        y_s[wave][p] = (v[j] - mu) * rstd * ln_g[p] + ln_b[p];
    }
    __syncthreads();
    if (lane < 6) {
        float a = outcb[lane];
        const float* w = outcW + lane * 128;
        for (int k = 0; k < 128; ++k) a += y_s[wave][k] * w[k];
        cmd_s[wave][lane] = a;
        out[(size_t)idx * 6 + lane] = a;
    }
    __syncthreads();
    if (lane < 6) {
        float a = outxb[lane];
        const float* w = outxW + lane * 134;
        for (int k = 0; k < 128; ++k) a += y_s[wave][k] * w[k];
        for (int j = 0; j < 6; ++j)  a += cmd_s[wave][j] * w[128 + j];
        out[(size_t)(B_ * S_ * 6) + (size_t)idx * 6 + lane] = tanhf(a * scale_p[0]);
    }
}

// ---------------------------------------------------------------------------
extern "C" void kernel_launch(void* const* d_in, const int* in_sizes, int n_in,
                              void* d_out, int out_size, void* d_ws, size_t ws_size,
                              hipStream_t stream) {
    const float* x      = (const float*)d_in[0];
    const float* ctx    = (const float*)d_in[1];
    const float* cmdW   = (const float*)d_in[2];
    const float* cmdb   = (const float*)d_in[3];
    const float* coordW = (const float*)d_in[4];
    const float* coordb = (const float*)d_in[5];
    const float* W_ih   = (const float*)d_in[6];
    const float* W_hh   = (const float*)d_in[7];
    const float* b_ih   = (const float*)d_in[8];
    const float* b_hh   = (const float*)d_in[9];
    const float* W_hr   = (const float*)d_in[10];
    const float* ln_g   = (const float*)d_in[11];
    const float* ln_b   = (const float*)d_in[12];
    const float* outcW  = (const float*)d_in[13];
    const float* outcb  = (const float*)d_in[14];
    const float* outxW  = (const float*)d_in[15];
    const float* outxb  = (const float*)d_in[16];
    const float* cscale = (const float*)d_in[17];

    char* ws = (char*)d_ws;
    float*          weff   = (float*)(ws + 0);            //  2048*16*4   = 128 KiB
    float*          base   = (float*)(ws + 131072);       //  64*2048*4   = 512 KiB
    unsigned short* whh_bf = (unsigned short*)(ws + 655360);   // 512 KiB
    unsigned short* whr_bf = (unsigned short*)(ws + 1179648);  // 128 KiB
    float*          hs     = (float*)(ws + 1310720);      //  64*512*128*4 = 16 MiB
    float*          xgf    = (float*)(ws + 18087936);     //  512*64*2048*4 = 256 MiB

    prep_weff<<<128, 256, 0, stream>>>(W_ih, cmdW, coordW, weff);
    prep_base<<<512, 256, 0, stream>>>(W_ih, ctx, cmdb, coordb, b_ih, b_hh, base);
    prep_cvt<<<1280, 256, 0, stream>>>(W_hh, W_hr, whh_bf, whr_bf);
    xgates_kernel<<<dim3(128, 4, 512), 256, 0, stream>>>(x, weff, base, xgf);
    scan_kernel<<<4, 256, 0, stream>>>(xgf, whh_bf, whr_bf, hs);
    out_kernel<<<4096, 256, 0, stream>>>(hs, ln_g, ln_b, outcW, outcb,
                                         outxW, outxb, cscale, (float*)d_out);
}
